// SimpleTransformerFP4_11003706212995
// MI455X (gfx1250) — matrix operand order repaired and checked by reading, not compiled, not run on hardware
//
#include <hip/hip_runtime.h>
#include <hip/hip_bf16.h>

// ---------------------------------------------------------------------------
// Problem constants (from reference): B=4, S=2048, H=1024, NH=8, D=128, L=6
// ---------------------------------------------------------------------------
#define TL   6
#define TH   1024
#define TNH  8
#define TD   128
#define TB   4
#define TS   2048
#define TM   (TB * TS)          // 8192 rows of activations

typedef __attribute__((ext_vector_type(8)))  __bf16 v8bf;
typedef __attribute__((ext_vector_type(16))) __bf16 v16bf;
typedef __attribute__((ext_vector_type(8)))  float  v8f;

// ---------------------------------------------------------------------------
// WMMA: D = A(16x32 bf16) * B(32x16 bf16) + C(16x16 f32)
// ---------------------------------------------------------------------------
__device__ __forceinline__ v8f wmma_bf16(v16bf a, v16bf b, v8f c) {
#if defined(__HIP_DEVICE_COMPILE__)
  return __builtin_amdgcn_wmma_f32_16x16x32_bf16(
      /*neg_a=*/false, a, /*neg_b=*/false, b,
      /*c_mod=*/(short)0, c, /*reuse_a=*/false, /*reuse_b=*/false);
#else
  (void)a; (void)b; return c;
#endif
}

// A-matrix fragment (16x32, lane m = lane%16, hl = lane/16):
//   lane holds K = hl*16 .. hl*16+15  (two 16B blocks)
__device__ __forceinline__ v16bf load_fragA(const __bf16* row, int kbase, int hl) {
  v8bf lo = *reinterpret_cast<const v8bf*>(row + kbase + hl * 16);
  v8bf hi = *reinterpret_cast<const v8bf*>(row + kbase + hl * 16 + 8);
  v16bf r;
#pragma unroll
  for (int i = 0; i < 8; ++i) { r[i] = lo[i]; r[8 + i] = hi[i]; }
  return r;
}

// B-matrix fragment (32x16, lane n = lane%16, hl = lane/16):
//   lane holds K = hl*16 .. hl*16+15 (16 consecutive bf16 = 32B); caller
//   passes pointer already offset by hl*16.
__device__ __forceinline__ v16bf load_fragB(const __bf16* p) {
  v8bf lo = *reinterpret_cast<const v8bf*>(p);
  v8bf hi = *reinterpret_cast<const v8bf*>(p + 8);
  v16bf r;
#pragma unroll
  for (int i = 0; i < 8; ++i) { r[i] = lo[i]; r[8 + i] = hi[i]; }
  return r;
}

// ---------------------------------------------------------------------------
// LayerNorm: one 256-thread block per row of H=1024
// ---------------------------------------------------------------------------
__global__ __launch_bounds__(256) void layernorm_k(
    const __bf16* __restrict__ in, const __bf16* __restrict__ g,
    const __bf16* __restrict__ b, __bf16* __restrict__ out) {
  __shared__ float red[256];
  const int row = blockIdx.x;
  const int t = threadIdx.x;
  const __bf16* xr = in + (size_t)row * TH;

  float vals[4];
  float s = 0.0f;
#pragma unroll
  for (int i = 0; i < 4; ++i) { vals[i] = (float)xr[t + 256 * i]; s += vals[i]; }
  red[t] = s; __syncthreads();
  for (int off = 128; off > 0; off >>= 1) {
    if (t < off) red[t] += red[t + off];
    __syncthreads();
  }
  const float mu = red[0] * (1.0f / TH);
  __syncthreads();

  float s2 = 0.0f;
#pragma unroll
  for (int i = 0; i < 4; ++i) { float d = vals[i] - mu; s2 += d * d; }
  red[t] = s2; __syncthreads();
  for (int off = 128; off > 0; off >>= 1) {
    if (t < off) red[t] += red[t + off];
    __syncthreads();
  }
  const float rstd = rsqrtf(red[0] * (1.0f / TH) + 1e-5f);

  __bf16* orow = out + (size_t)row * TH;
#pragma unroll
  for (int i = 0; i < 4; ++i) {
    int col = t + 256 * i;
    float y = (vals[i] - mu) * rstd * (float)g[col] + (float)b[col];
    orow[col] = (__bf16)y;
  }
}

// ---------------------------------------------------------------------------
// GEMM: out[M,N] = act[M,K] * W[N,K]^T + bias[N]  (+residual, +ReLU)
// Both operands K-contiguous.
// 8 waves/block arranged 4x2; wave tile 32x64 (2x4 WMMA tiles); block 128x128.
// Software-pipelined: fragments for k+32 are fetched before the WMMAs for k,
// so global loads overlap matrix math instead of serializing behind waits.
// ---------------------------------------------------------------------------
#define GEMM_LOAD(FA0, FA1, FB0, FB1, FB2, FB3, KK)    \
  do {                                                 \
    FA0 = load_fragA(a0, (KK), hl);                    \
    FA1 = load_fragA(a1, (KK), hl);                    \
    FB0 = load_fragB(wp0 + (KK));                      \
    FB1 = load_fragB(wp1 + (KK));                      \
    FB2 = load_fragB(wp2 + (KK));                      \
    FB3 = load_fragB(wp3 + (KK));                      \
  } while (0)

#define GEMM_MATH(FA0, FA1, FB0, FB1, FB2, FB3)        \
  do {                                                 \
    acc00 = wmma_bf16(FA0, FB0, acc00);                \
    acc01 = wmma_bf16(FA0, FB1, acc01);                \
    acc02 = wmma_bf16(FA0, FB2, acc02);                \
    acc03 = wmma_bf16(FA0, FB3, acc03);                \
    acc10 = wmma_bf16(FA1, FB0, acc10);                \
    acc11 = wmma_bf16(FA1, FB1, acc11);                \
    acc12 = wmma_bf16(FA1, FB2, acc12);                \
    acc13 = wmma_bf16(FA1, FB3, acc13);                \
  } while (0)

template <bool RELU, bool RES>
__global__ __launch_bounds__(256) void gemm_wmma_k(
    const __bf16* __restrict__ A, const __bf16* __restrict__ W,
    const __bf16* __restrict__ bias, const __bf16* __restrict__ res,
    __bf16* __restrict__ out, int N, int K) {
  const int w = threadIdx.x >> 5, lane = threadIdx.x & 31;
  const int ln = lane & 15, hl = lane >> 4;
  const int wm = w >> 1, wn = w & 1;              // 4 x 2 waves
  const int m0 = blockIdx.x * 128 + wm * 32;
  const int n0 = blockIdx.y * 128 + wn * 64;

  v8f acc00 = {}, acc01 = {}, acc02 = {}, acc03 = {};
  v8f acc10 = {}, acc11 = {}, acc12 = {}, acc13 = {};

  const __bf16* a0 = A + (size_t)(m0 + ln) * K;
  const __bf16* a1 = a0 + (size_t)16 * K;
  const __bf16* wp0 = W + (size_t)(n0 + ln) * K + hl * 16;
  const __bf16* wp1 = wp0 + (size_t)16 * K;
  const __bf16* wp2 = wp0 + (size_t)32 * K;
  const __bf16* wp3 = wp0 + (size_t)48 * K;

  // double-buffered fragment registers
  v16bf xa0, xa1, xb0, xb1, xb2, xb3;   // buffer X (even k-steps)
  v16bf ya0, ya1, yb0, yb1, yb2, yb3;   // buffer Y (odd  k-steps)

  GEMM_LOAD(xa0, xa1, xb0, xb1, xb2, xb3, 0);
  for (int k = 0; k + 64 <= K; k += 64) {       // K % 64 == 0 always here
    GEMM_LOAD(ya0, ya1, yb0, yb1, yb2, yb3, k + 32);
    GEMM_MATH(xa0, xa1, xb0, xb1, xb2, xb3);
    if (k + 64 < K) GEMM_LOAD(xa0, xa1, xb0, xb1, xb2, xb3, k + 64);
    GEMM_MATH(ya0, ya1, yb0, yb1, yb2, yb3);
  }

  v8f accs[2][4] = {{acc00, acc01, acc02, acc03}, {acc10, acc11, acc12, acc13}};
#pragma unroll
  for (int mt = 0; mt < 2; ++mt) {
#pragma unroll
    for (int nt = 0; nt < 4; ++nt) {
      const int col = n0 + nt * 16 + ln;
      const float bv = (float)bias[col];
      v8f a = accs[mt][nt];
#pragma unroll
      for (int v = 0; v < 8; ++v) {
        const int row = m0 + mt * 16 + 8 * hl + v;
        float xv = a[v] + bv;
        if (RES) xv += (float)res[(size_t)row * N + col];
        if (RELU) xv = fmaxf(xv, 0.0f);
        out[(size_t)row * N + col] = (__bf16)xv;
      }
    }
  }
}

// ---------------------------------------------------------------------------
// Flash attention: one block per (batch*head, 64-query tile).
// Score phase: 8 waves as 4x2 (16q x 32k each). PV phase: wave w owns the
// 16-wide d-slice [w*16, w*16+16) for all 64 queries.
// ---------------------------------------------------------------------------
__global__ __launch_bounds__(256) void attn_k(const __bf16* __restrict__ qkv,
                                              __bf16* __restrict__ ao) {
  const int qt = blockIdx.x;              // 0..31 (query tile)
  const int bh = blockIdx.y;              // 0..31
  const int bb = bh >> 3, hh = bh & 7;
  const int q0 = qt * 64;
  const int w = threadIdx.x >> 5, lane = threadIdx.x & 31;
  const int ln = lane & 15, hl = lane >> 4;
  const int wm = w >> 1, wn = w & 1;
  const int rs = 3 * TH;                  // qkv row stride (elements)

  const __bf16* Qb = qkv + (size_t)bb * TS * rs + hh * TD;
  const __bf16* Kb = Qb + TH;
  const __bf16* Vb = Qb + 2 * TH;

  __shared__ float  Sf[64 * 64];          // raw scores (f32)
  __shared__ __bf16 Pm[64 * 64];          // exp probabilities (bf16)
  __shared__ __bf16 Vt[TD * 64];          // transposed V chunk: Vt[d][k]
  __shared__ float  rowm[64], rowl[64], rowf[64];

  // Q fragments for this wave's 16 query rows, all of D=128 (4 k-steps)
  v16bf qf[4];
  {
    const __bf16* qrow = Qb + (size_t)(q0 + wm * 16 + ln) * rs;
#pragma unroll
    for (int ks = 0; ks < 4; ++ks) qf[ks] = load_fragA(qrow, ks * 32, hl);
  }

  v8f oacc[4] = {};                       // 64q x 16d output accumulator
  if (threadIdx.x < 64) { rowm[threadIdx.x] = -1e30f; rowl[threadIdx.x] = 0.0f; }
  __syncthreads();

  const float scale = 0.08838834764831845f;  // 1/sqrt(128)

  for (int kc = 0; kc < TS; kc += 64) {
    // ---- scores: Q(16x128) . K^T(128x32) for this wave ----
    v8f sacc[2] = {};
#pragma unroll
    for (int ks = 0; ks < 4; ++ks) {
      const __bf16* k0 = Kb + (size_t)(kc + wn * 32 + ln) * rs + ks * 32 + hl * 16;
      const __bf16* k1 = k0 + (size_t)16 * rs;
      v16bf fb0 = load_fragB(k0);
      v16bf fb1 = load_fragB(k1);
      sacc[0] = wmma_bf16(qf[ks], fb0, sacc[0]);
      sacc[1] = wmma_bf16(qf[ks], fb1, sacc[1]);
    }
#pragma unroll
    for (int nt = 0; nt < 2; ++nt)
#pragma unroll
      for (int v = 0; v < 8; ++v)
        Sf[(wm * 16 + 8 * hl + v) * 64 + wn * 32 + nt * 16 + ln] = sacc[nt][v] * scale;

    // ---- stage V chunk transposed into LDS ----
    for (int g = threadIdx.x; g < 64 * 16; g += 256) {
      const int kk = g >> 4, d8 = (g & 15) * 8;
      v8bf vv = *reinterpret_cast<const v8bf*>(Vb + (size_t)(kc + kk) * rs + d8);
#pragma unroll
      for (int j = 0; j < 8; ++j) Vt[(d8 + j) * 64 + kk] = vv[j];
    }
    __syncthreads();

    // ---- online softmax per query row ----
    if (threadIdx.x < 64) {
      const int r = threadIdx.x;
      const float mo = rowm[r];
      float mx = mo;
      for (int j = 0; j < 64; ++j) mx = fmaxf(mx, Sf[r * 64 + j]);
      const float f = __expf(mo - mx);
      float sum = 0.0f;
      for (int j = 0; j < 64; ++j) {
        float p = __expf(Sf[r * 64 + j] - mx);
        Pm[r * 64 + j] = (__bf16)p;
        sum += p;
      }
      rowl[r] = rowl[r] * f + sum;
      rowm[r] = mx;
      rowf[r] = f;
    }
    __syncthreads();

    // ---- rescale O, then O += P(64x64) . V(64x16-slice) ----
#pragma unroll
    for (int tm = 0; tm < 4; ++tm)
#pragma unroll
      for (int v = 0; v < 8; ++v)
        oacc[tm][v] *= rowf[tm * 16 + 8 * hl + v];

#pragma unroll
    for (int kt = 0; kt < 2; ++kt) {
      v16bf vb = load_fragB(&Vt[(w * 16 + ln) * 64 + kt * 32 + hl * 16]);
#pragma unroll
      for (int tm = 0; tm < 4; ++tm) {
        v16bf pa = load_fragA(&Pm[(tm * 16 + ln) * 64], kt * 32, hl);
        oacc[tm] = wmma_bf16(pa, vb, oacc[tm]);
      }
    }
    __syncthreads();
  }

  // ---- finalize: divide by row sum, write ao[b, q, h*D + d] ----
#pragma unroll
  for (int tm = 0; tm < 4; ++tm)
#pragma unroll
    for (int v = 0; v < 8; ++v) {
      const int r = tm * 16 + 8 * hl + v;
      const float val = oacc[tm][v] / rowl[r];
      ao[((size_t)bb * TS + q0 + r) * TH + hh * TD + w * 16 + ln] = (__bf16)val;
    }
}

// ---------------------------------------------------------------------------
// Host-side orchestration
// ---------------------------------------------------------------------------
extern "C" void kernel_launch(void* const* d_in, const int* in_sizes, int n_in,
                              void* d_out, int out_size, void* d_ws, size_t ws_size,
                              hipStream_t stream) {
  (void)in_sizes; (void)n_in; (void)out_size; (void)ws_size;
  const __bf16* x      = (const __bf16*)d_in[0];
  const __bf16* qkv_w  = (const __bf16*)d_in[1];
  const __bf16* qkv_b  = (const __bf16*)d_in[2];
  const __bf16* proj_w = (const __bf16*)d_in[3];
  const __bf16* proj_b = (const __bf16*)d_in[4];
  const __bf16* fc1_w  = (const __bf16*)d_in[5];
  const __bf16* fc1_b  = (const __bf16*)d_in[6];
  const __bf16* fc2_w  = (const __bf16*)d_in[7];
  const __bf16* fc2_b  = (const __bf16*)d_in[8];
  const __bf16* n1_g   = (const __bf16*)d_in[9];
  const __bf16* n1_b   = (const __bf16*)d_in[10];
  const __bf16* n2_g   = (const __bf16*)d_in[11];
  const __bf16* n2_b   = (const __bf16*)d_in[12];

  char* ws = (char*)d_ws;
  __bf16* xcur = (__bf16*)ws;                          // 16 MB running x
  __bf16* hbuf = (__bf16*)(ws + ((size_t)16 << 20));   // 16 MB: ln-out / attn-out
  __bf16* big  = (__bf16*)(ws + ((size_t)32 << 20));   // 64 MB: qkv / fc1-out

  const dim3 blk(256);
  for (int l = 0; l < TL; ++l) {
    const __bf16* xin = (l == 0) ? x : xcur;

    // --- attention block ---
    layernorm_k<<<TM, blk, 0, stream>>>(xin, n1_g + l * TH, n1_b + l * TH, hbuf);
    gemm_wmma_k<false, false><<<dim3(TM / 128, (3 * TH) / 128), blk, 0, stream>>>(
        hbuf, qkv_w + (size_t)l * 3 * TH * TH, qkv_b + (size_t)l * 3 * TH,
        nullptr, big, 3 * TH, TH);
    attn_k<<<dim3(TS / 64, TB * TNH), blk, 0, stream>>>(big, hbuf);
    gemm_wmma_k<false, true><<<dim3(TM / 128, TH / 128), blk, 0, stream>>>(
        hbuf, proj_w + (size_t)l * TH * TH, proj_b + (size_t)l * TH,
        xin, xcur, TH, TH);

    // --- FFN block ---
    layernorm_k<<<TM, blk, 0, stream>>>(xcur, n2_g + l * TH, n2_b + l * TH, hbuf);
    gemm_wmma_k<true, false><<<dim3(TM / 128, (4 * TH) / 128), blk, 0, stream>>>(
        hbuf, fc1_w + (size_t)l * 4 * TH * TH, fc1_b + (size_t)l * 4 * TH,
        nullptr, big, 4 * TH, TH);
    __bf16* outp = (l == TL - 1) ? (__bf16*)d_out : xcur;
    gemm_wmma_k<false, true><<<dim3(TM / 128, TH / 128), blk, 0, stream>>>(
        big, fc2_w + (size_t)l * TH * 4 * TH, fc2_b + (size_t)l * TH,
        xcur, outp, TH, 4 * TH);
  }
}
